// GNNRetrievalModel_45561013076584
// MI455X (gfx1250) — matrix operand order repaired and checked by reading, not compiled, not run on hardware
//
#include <hip/hip_runtime.h>
#include <hip/hip_bf16.h>

// ---------------- CDNA5 WMMA types ----------------
typedef __attribute__((ext_vector_type(16))) __bf16 v16bf;
typedef __attribute__((ext_vector_type(8)))  __bf16 v8bf;
typedef __attribute__((ext_vector_type(8)))  float  v8f;

#define NNODES 100000
#define NEDGES 500000
#define NREL   4
#define DIN    256
#define HDIM   128
#define LNEPS  1e-5f

__device__ __forceinline__ unsigned short f2bf_bits(float f) {
    unsigned int u = __float_as_uint(f);
    u += 0x7FFFu + ((u >> 16) & 1u);   // round-to-nearest-even
    return (unsigned short)(u >> 16);
}

// ---------------- utility kernels ----------------
__global__ __launch_bounds__(256) void k_zero_u32(unsigned int* p, long long n) {
    long long i = (long long)blockIdx.x * blockDim.x + threadIdx.x;
    long long stride = (long long)gridDim.x * blockDim.x;
    for (; i < n; i += stride) p[i] = 0u;
}

__global__ __launch_bounds__(256) void k_f32_to_bf16(const float* __restrict__ in,
                                                     unsigned short* __restrict__ out,
                                                     long long n) {
    long long i = (long long)blockIdx.x * blockDim.x + threadIdx.x;
    long long stride = (long long)gridDim.x * blockDim.x;
    for (; i < n; i += stride) out[i] = f2bf_bits(in[i]);
}

// ---------------- weight packing into WMMA B-fragment layout ----------------
// pack[((kt*NT + nt)*32 + lane)*16 + e]  holds  Wcat[K = kt*32 + 16*(lane>>4) + e][col = nt*16 + (lane&15)]
// so each lane loads its whole 16-element B fragment as one contiguous 32-byte v16bf.
__global__ __launch_bounds__(256) void k_pack_w0(const float* __restrict__ Wrel,   // [4,256,128]
                                                 const float* __restrict__ Wroot,  // [256,128]
                                                 const float* __restrict__ Wproj,  // [256,128]
                                                 unsigned short* __restrict__ pack) {
    const int NT = 48;                      // 768 cols / 16
    int p = blockIdx.x * 256 + threadIdx.x;
    if (p >= 8 * NT * 512) return;          // KT=8 ktiles * NT ntiles * 512 elems
    int e    = p & 15;
    int lane = (p >> 4) & 31;
    int tile = p >> 9;                      // kt*NT + nt
    int nt   = tile % NT;
    int kt   = tile / NT;
    int K    = kt * 32 + ((lane >> 4) << 4) + e;
    int col  = nt * 16 + (lane & 15);
    float v;
    if (col < 512) { int r = col >> 7, f = col & 127; v = Wrel[((size_t)r * DIN + K) * HDIM + f]; }
    else if (col < 640) v = Wroot[(size_t)K * HDIM + (col - 512)];
    else                v = Wproj[(size_t)K * HDIM + (col - 640)];
    pack[p] = f2bf_bits(v);
}

__global__ __launch_bounds__(256) void k_pack_w1(const float* __restrict__ Wrel,   // [4,128,128]
                                                 const float* __restrict__ Wroot,  // [128,128]
                                                 unsigned short* __restrict__ pack) {
    const int NT = 40;                      // 640 cols / 16
    int p = blockIdx.x * 256 + threadIdx.x;
    if (p >= 4 * NT * 512) return;          // KT=4
    int e    = p & 15;
    int lane = (p >> 4) & 31;
    int tile = p >> 9;
    int nt   = tile % NT;
    int kt   = tile / NT;
    int K    = kt * 32 + ((lane >> 4) << 4) + e;
    int col  = nt * 16 + (lane & 15);
    float v;
    if (col < 512) { int r = col >> 7, f = col & 127; v = Wrel[((size_t)r * HDIM + K) * HDIM + f]; }
    else           v = Wroot[(size_t)K * HDIM + (col - 512)];
    pack[p] = f2bf_bits(v);
}

// ---------------- bf16 WMMA GEMM: C[M x (NT*16)] = A[M x (KT*32)] * Bpack ----------------
// One wave computes a 16(M) x 64(N) strip (4 WMMA tiles), reusing its A fragment 4x.
__global__ __launch_bounds__(256) void k_gemm_bf16_wmma(const unsigned short* __restrict__ A,
                                                        const unsigned short* __restrict__ B,
                                                        float* __restrict__ C,
                                                        int M, int KT, int NT, int lda, int ldc) {
    int w    = (int)((blockIdx.x * blockDim.x + threadIdx.x) >> 5);
    int lane = threadIdx.x & 31;
    int NG   = NT >> 2;                 // groups of 4 n-tiles
    int MT   = M >> 4;
    if (w >= MT * NG) return;
    int ng   = w % NG;
    int mt   = w / NG;
    int half = lane >> 4;
    int lm   = lane & 15;

    // A layout (matches B pack): lane holds row M=lm; elems 0-15 = K(kb + 16*half .. +15)
    //  -> two contiguous 16B loads.
    const unsigned short* arow  = A + (size_t)(mt * 16 + lm) * lda + half * 16;
    const unsigned short* bbase = B + ((size_t)(ng * 4) * 32 + lane) * 16;

    v8f c0 = {}, c1 = {}, c2 = {}, c3 = {};
    for (int kt = 0; kt < KT; ++kt) {
        v8bf lo = *(const v8bf*)(arow + kt * 32);
        v8bf hi = *(const v8bf*)(arow + kt * 32 + 8);
        v16bf a;
#pragma unroll
        for (int i = 0; i < 8; ++i) { a[i] = lo[i]; a[8 + i] = hi[i]; }
        const unsigned short* bk = bbase + (size_t)kt * NT * 512;
        v16bf b0 = *(const v16bf*)(bk);
        v16bf b1 = *(const v16bf*)(bk + 512);
        v16bf b2 = *(const v16bf*)(bk + 1024);
        v16bf b3 = *(const v16bf*)(bk + 1536);
        c0 = __builtin_amdgcn_wmma_f32_16x16x32_bf16(false, a, false, b0, (short)0, c0, false, false);
        c1 = __builtin_amdgcn_wmma_f32_16x16x32_bf16(false, a, false, b1, (short)0, c1, false, false);
        c2 = __builtin_amdgcn_wmma_f32_16x16x32_bf16(false, a, false, b2, (short)0, c2, false, false);
        c3 = __builtin_amdgcn_wmma_f32_16x16x32_bf16(false, a, false, b3, (short)0, c3, false, false);
    }
    // C layout: lane (0-15) -> N=lane, VGPR j -> M=j ; lanes 16-31 -> M=j+8
    float* crow = C + (size_t)(mt * 16 + half * 8) * ldc + ng * 64 + lm;
#pragma unroll
    for (int j = 0; j < 8; ++j) {
        size_t o = (size_t)j * ldc;
        crow[o]      = c0[j];
        crow[o + 16] = c1[j];
        crow[o + 32] = c2[j];
        crow[o + 48] = c3[j];
    }
}

// ---------------- edge kernels ----------------
__global__ __launch_bounds__(256) void k_count(const int* __restrict__ dst,
                                               const int* __restrict__ et,
                                               int* __restrict__ cnt, int E) {
    int e = blockIdx.x * 256 + threadIdx.x;
    if (e < E) atomicAdd(&cnt[dst[e] * NREL + et[e]], 1);
}

__global__ __launch_bounds__(256) void k_scatter(const int* __restrict__ src,
                                                 const int* __restrict__ dst,
                                                 const int* __restrict__ et,
                                                 const float* __restrict__ msg, int ldm,
                                                 float* __restrict__ agg, int E) {
    long long g = (long long)blockIdx.x * 256 + threadIdx.x;
    int e = (int)(g >> 7);
    if (e >= E) return;
    int h = (int)(g & 127);
    int s = src[e], d = dst[e], t = et[e];
    float v = msg[(size_t)s * ldm + t * HDIM + h];
    atomicAdd(&agg[((size_t)d * NREL + t) * HDIM + h], v);
}

// ---------------- per-node epilogue: mean + root + relu + residual + LayerNorm ----------------
// One wave32 per node, 4 channels per lane, shfl reductions for mean/var.
__global__ __launch_bounds__(256) void k_finalize(const float* __restrict__ agg,
                                                  const int* __restrict__ cnt,
                                                  const float* __restrict__ G, int ldg, int rootcol,
                                                  const float* __restrict__ b,
                                                  const float* __restrict__ bproj,   // layer0 only
                                                  const float* __restrict__ res,     // layer1 only
                                                  const float* __restrict__ gamma,
                                                  const float* __restrict__ beta,
                                                  float* __restrict__ outf,
                                                  unsigned short* __restrict__ outbf, // layer0 only
                                                  int n_nodes) {
    int wid  = (int)((blockIdx.x * blockDim.x + threadIdx.x) >> 5);
    int lane = threadIdx.x & 31;
    if (wid >= n_nodes) return;

    const float* aggn = agg + (size_t)wid * NREL * HDIM;
    float inv[NREL];
#pragma unroll
    for (int r = 0; r < NREL; ++r) {
        int c = cnt[wid * NREL + r];
        inv[r] = 1.0f / fmaxf((float)c, 1.0f);
    }
    const float* grow = G + (size_t)wid * ldg;

    float v[4];
    float s1 = 0.f, s2 = 0.f;
#pragma unroll
    for (int j = 0; j < 4; ++j) {
        int h = j * 32 + lane;
        float sum = 0.f;
#pragma unroll
        for (int r = 0; r < NREL; ++r) sum += aggn[r * HDIM + h] * inv[r];
        float pre = sum + grow[rootcol + h] + b[h];
        float hv  = fmaxf(pre, 0.f);
        if (bproj) hv += grow[rootcol + HDIM + h] + bproj[h];
        if (res)   hv += res[(size_t)wid * HDIM + h];
        v[j] = hv;
        s1 += hv;
        s2 += hv * hv;
    }
#pragma unroll
    for (int off = 16; off > 0; off >>= 1) {
        s1 += __shfl_xor(s1, off);
        s2 += __shfl_xor(s2, off);
    }
    float mu  = s1 * (1.0f / HDIM);
    float var = s2 * (1.0f / HDIM) - mu * mu;
    float rs  = rsqrtf(var + LNEPS);
#pragma unroll
    for (int j = 0; j < 4; ++j) {
        int h = j * 32 + lane;
        float o = (v[j] - mu) * rs * gamma[h] + beta[h];
        outf[(size_t)wid * HDIM + h] = o;
        if (outbf) outbf[(size_t)wid * HDIM + h] = f2bf_bits(o);
    }
}

// ---------------- host launch ----------------
extern "C" void kernel_launch(void* const* d_in, const int* in_sizes, int n_in,
                              void* d_out, int out_size, void* d_ws, size_t ws_size,
                              hipStream_t stream) {
    (void)in_sizes; (void)n_in; (void)out_size; (void)ws_size;
    const float* x       = (const float*)d_in[0];
    const int*   eidx    = (const int*)d_in[1];     // [2,E]: src then dst
    const int*   etype   = (const int*)d_in[2];
    const float* W_rel0  = (const float*)d_in[3];
    const float* W_root0 = (const float*)d_in[4];
    const float* b0      = (const float*)d_in[5];
    const float* W_rel1  = (const float*)d_in[6];
    const float* W_root1 = (const float*)d_in[7];
    const float* b1      = (const float*)d_in[8];
    const float* W_proj  = (const float*)d_in[9];
    const float* b_proj  = (const float*)d_in[10];
    const float* gamma0  = (const float*)d_in[11];
    const float* beta0   = (const float*)d_in[12];
    const float* gamma1  = (const float*)d_in[13];
    const float* beta1   = (const float*)d_in[14];
    const int* src = eidx;
    const int* dst = eidx + NEDGES;

    // workspace carve-up (≈640 MB peak)
    char* base = (char*)d_ws;
    size_t off = 0;
    auto take = [&](size_t bytes) -> char* {
        char* p = base + off;
        off += (bytes + 255) & ~(size_t)255;
        return p;
    };
    unsigned short* xbf   = (unsigned short*)take((size_t)NNODES * DIN * 2);
    unsigned short* hbf   = (unsigned short*)take((size_t)NNODES * HDIM * 2);
    unsigned short* pack0 = (unsigned short*)take((size_t)8 * 48 * 512 * 2);
    unsigned short* pack1 = (unsigned short*)take((size_t)4 * 40 * 512 * 2);
    float* gemm_out = (float*)take((size_t)NNODES * 768 * 4);   // layer0: 768 cols; layer1 reuses (640 cols)
    float* agg      = (float*)take((size_t)NNODES * NREL * HDIM * 4);
    int*   cnt      = (int*)take((size_t)NNODES * NREL * 4);
    float* hbuf     = (float*)take((size_t)NNODES * HDIM * 4);

    // ---- layer 0 ----
    k_zero_u32<<<4096, 256, 0, stream>>>((unsigned int*)agg, (long long)NNODES * NREL * HDIM);
    k_zero_u32<<<1024, 256, 0, stream>>>((unsigned int*)cnt, (long long)NNODES * NREL);
    k_f32_to_bf16<<<4096, 256, 0, stream>>>(x, xbf, (long long)NNODES * DIN);
    k_pack_w0<<<(8 * 48 * 512 + 255) / 256, 256, 0, stream>>>(W_rel0, W_root0, W_proj, pack0);
    k_pack_w1<<<(4 * 40 * 512 + 255) / 256, 256, 0, stream>>>(W_rel1, W_root1, pack1);

    {   // [100000 x 256] * [256 x 768] (rel0 | root0 | proj)
        int waves = (NNODES / 16) * (48 / 4);
        k_gemm_bf16_wmma<<<(waves + 7) / 8, 256, 0, stream>>>(xbf, pack0, gemm_out,
                                                              NNODES, 8, 48, DIN, 768);
    }
    k_count<<<(NEDGES + 255) / 256, 256, 0, stream>>>(dst, etype, cnt, NEDGES);
    k_scatter<<<(int)(((long long)NEDGES * 128) / 256), 256, 0, stream>>>(src, dst, etype,
                                                                          gemm_out, 768, agg, NEDGES);
    k_finalize<<<(NNODES + 7) / 8, 256, 0, stream>>>(agg, cnt, gemm_out, 768, 512,
                                                     b0, b_proj, nullptr,
                                                     gamma0, beta0, hbuf, hbf, NNODES);

    // ---- layer 1 ----
    k_zero_u32<<<4096, 256, 0, stream>>>((unsigned int*)agg, (long long)NNODES * NREL * HDIM);
    {   // [100000 x 128] * [128 x 640] (rel1 | root1)
        int waves = (NNODES / 16) * (40 / 4);
        k_gemm_bf16_wmma<<<(waves + 7) / 8, 256, 0, stream>>>(hbf, pack1, gemm_out,
                                                              NNODES, 4, 40, HDIM, 640);
    }
    k_scatter<<<(int)(((long long)NEDGES * 128) / 256), 256, 0, stream>>>(src, dst, etype,
                                                                          gemm_out, 640, agg, NEDGES);
    k_finalize<<<(NNODES + 7) / 8, 256, 0, stream>>>(agg, cnt, gemm_out, 640, 512,
                                                     b1, nullptr, hbuf,
                                                     gamma1, beta1, (float*)d_out, nullptr, NNODES);
}
